// LSTMLayer_61521111548505
// MI455X (gfx1250) — matrix operand order repaired and checked by reading, not compiled, not run on hardware
//
#include <hip/hip_runtime.h>
#include <hip/hip_bf16.h>
#include <cstdint>
#include <cstddef>

#define B_  64
#define T_  512
#define I_  1024
#define H_  1024
#define NG_ 4096   // 4*H
#define KS_ 4      // K-splits for the recurrent GEMM

typedef __attribute__((ext_vector_type(16))) __bf16 v16bf;
typedef __attribute__((ext_vector_type(8)))  float  v8f;

union Frag {
  v16bf v;
  unsigned int d[8];
  unsigned short u[16];
};

__device__ __forceinline__ unsigned short f2bf(float f) {
  unsigned int u = __float_as_uint(f);
  u += 0x7FFFu + ((u >> 16) & 1u);   // round-to-nearest-even
  return (unsigned short)(u >> 16);
}

// A matrix 16x32 bf16, row-major source A[row][k] with leading dim lda.
// CDNA5 layout: lanes 0-15 -> M=0..15 (lanes 16-31 same M, K offset +8);
// VGPR 0..3 hold K = 2v,2v+1 (+half*8); VGPR 4..7 hold K = 16+2(v-4),+1 (+half*8).
// The two 8-ushort runs are contiguous -> compiler merges into 2x global_load_b128.
__device__ __forceinline__ void load_a_frag(Frag& fa, const unsigned short* __restrict__ A,
                                            int lda, int row0, int k0, int lane) {
  const int m    = lane & 15;
  const int half = lane >> 4;
  const unsigned short* base = A + (size_t)(row0 + m) * lda + k0 + half * 8;
  fa.d[0] = *(const unsigned int*)(base + 0);
  fa.d[1] = *(const unsigned int*)(base + 2);
  fa.d[2] = *(const unsigned int*)(base + 4);
  fa.d[3] = *(const unsigned int*)(base + 6);
  fa.d[4] = *(const unsigned int*)(base + 16);
  fa.d[5] = *(const unsigned int*)(base + 18);
  fa.d[6] = *(const unsigned int*)(base + 20);
  fa.d[7] = *(const unsigned int*)(base + 22);
}

// B matrix 32x16 bf16 where B[k][n] = W[col0+n][k0+k], W row-major [N][K], ld = ldw.
// CDNA5 layout: lane%16 -> N; elements 0..7 hold K = half*8+0..7, elements 8..15 hold K = 16+half*8+0..7.
__device__ __forceinline__ void load_b_frag(Frag& fb, const unsigned short* __restrict__ W,
                                            int ldw, int col0, int k0, int lane) {
  const int n    = lane & 15;
  const int koff = (lane >> 4) * 8;
  const uint4* p = (const uint4*)(W + (size_t)(col0 + n) * ldw + k0 + koff);
  uint4 t0 = p[0];
  uint4 t1 = p[2];
  fb.d[0] = t0.x; fb.d[1] = t0.y; fb.d[2] = t0.z; fb.d[3] = t0.w;
  fb.d[4] = t1.x; fb.d[5] = t1.y; fb.d[6] = t1.z; fb.d[7] = t1.w;
}

// ---------------- conversion / init kernels ----------------

__global__ __launch_bounds__(256) void cvt_bf16_kernel(const float4* __restrict__ src,
                                                       uint2* __restrict__ dst, int n4) {
  int i = blockIdx.x * 256 + threadIdx.x;
  if (i >= n4) return;
  float4 v = src[i];
  uint2 o;
  o.x = (unsigned)f2bf(v.x) | ((unsigned)f2bf(v.y) << 16);
  o.y = (unsigned)f2bf(v.z) | ((unsigned)f2bf(v.w) << 16);
  dst[i] = o;
}

__global__ __launch_bounds__(256) void init_state_kernel(const float* __restrict__ h0,
                                                         const float* __restrict__ c0,
                                                         unsigned short* __restrict__ hb,
                                                         float* __restrict__ c) {
  int i = blockIdx.x * 256 + threadIdx.x;   // B*H threads
  c[i]  = c0[i];
  hb[i] = f2bf(h0[i]);
}

__global__ __launch_bounds__(256) void bias_kernel(const float* __restrict__ bih,
                                                   const float* __restrict__ bhh,
                                                   float* __restrict__ bias) {
  int i = blockIdx.x * 256 + threadIdx.x;   // 4H threads
  bias[i] = bih[i] + bhh[i];
}

// ---------------- phase 1: G[t][b][:] = x[b][t][:] @ Wih^T + (bih+bhh) ----------------
// M = B*T (row = b*T + t), N = 4H, K = I. Wave: 16x64 strip (1 M-tile x 4 N-tiles).

__global__ __launch_bounds__(256) void gemm_x_kernel(
    const unsigned short* __restrict__ Xb,   // [B*T, I] bf16
    const unsigned short* __restrict__ Wb,   // [4H, I]  bf16
    const float* __restrict__ bias,          // [4H]
    float* __restrict__ G)                   // [T, B, 4H] fp32
{
  const int lane  = threadIdx.x & 31;
  const int wid   = threadIdx.x >> 5;
  const int mtile = blockIdx.x * 8 + wid;    // 0..2047
  const int row0  = mtile * 16;
  const int col0  = blockIdx.y * 64;         // 0..63 groups

  v8f acc0 = {}; v8f acc1 = {}; v8f acc2 = {}; v8f acc3 = {};
  for (int k0 = 0; k0 < I_; k0 += 32) {
    Frag a;  load_a_frag(a,  Xb, I_, row0, k0, lane);
    Frag b0; load_b_frag(b0, Wb, I_, col0 +  0, k0, lane);
    Frag b1; load_b_frag(b1, Wb, I_, col0 + 16, k0, lane);
    Frag b2; load_b_frag(b2, Wb, I_, col0 + 32, k0, lane);
    Frag b3; load_b_frag(b3, Wb, I_, col0 + 48, k0, lane);
    acc0 = __builtin_amdgcn_wmma_f32_16x16x32_bf16(false, a.v, false, b0.v, (short)0, acc0, false, false);
    acc1 = __builtin_amdgcn_wmma_f32_16x16x32_bf16(false, a.v, false, b1.v, (short)0, acc1, false, false);
    acc2 = __builtin_amdgcn_wmma_f32_16x16x32_bf16(false, a.v, false, b2.v, (short)0, acc2, false, false);
    acc3 = __builtin_amdgcn_wmma_f32_16x16x32_bf16(false, a.v, false, b3.v, (short)0, acc3, false, false);
  }

  const int nlo = lane & 15;
  const int rhi = (lane >> 4) * 8;
  #pragma unroll
  for (int r = 0; r < 8; ++r) {
    const int row = row0 + r + rhi;          // = b*T + t
    const int b   = row >> 9;                // / T_
    const int t   = row & (T_ - 1);
    const size_t gbase = ((size_t)t * B_ + b) * NG_;
    G[gbase + col0 +  0 + nlo] = acc0[r] + bias[col0 +  0 + nlo];
    G[gbase + col0 + 16 + nlo] = acc1[r] + bias[col0 + 16 + nlo];
    G[gbase + col0 + 32 + nlo] = acc2[r] + bias[col0 + 32 + nlo];
    G[gbase + col0 + 48 + nlo] = acc3[r] + bias[col0 + 48 + nlo];
  }
}

// ---------------- phase 2a: split-K partials of h @ Whh^T  (M=64, N=4H, K=H/KS_) --------
// 4 M-tiles x 64 N-groups x KS_ k-splits = 1024 wave-tiles = 128 blocks of 8 waves.
// Waves in a block share the N-group (same weight tiles -> WGP$ reuse).

__global__ __launch_bounds__(256) void gemm_h_kernel(
    const unsigned short* __restrict__ Hb,   // [B, H] bf16
    const unsigned short* __restrict__ Wb,   // [4H, H] bf16
    float* __restrict__ part)                // [KS_, B, 4H] fp32 partial sums
{
  const int lane = threadIdx.x & 31;
  const int wid  = threadIdx.x >> 5;
  const int gw   = blockIdx.x * 8 + wid;     // 0..1023
  const int ks    = gw & (KS_ - 1);          // k-split
  const int mtile = (gw >> 2) & 3;           // 4 M-tiles (B=64)
  const int ngrp  = gw >> 4;                 // 64 groups of 4 N-tiles
  const int row0 = mtile * 16;
  const int col0 = ngrp * 64;
  const int kbeg = ks * (H_ / KS_);
  const int kend = kbeg + (H_ / KS_);

  v8f acc0 = {}; v8f acc1 = {}; v8f acc2 = {}; v8f acc3 = {};
  for (int k0 = kbeg; k0 < kend; k0 += 32) {
    Frag a;  load_a_frag(a,  Hb, H_, row0, k0, lane);
    Frag b0; load_b_frag(b0, Wb, H_, col0 +  0, k0, lane);
    Frag b1; load_b_frag(b1, Wb, H_, col0 + 16, k0, lane);
    Frag b2; load_b_frag(b2, Wb, H_, col0 + 32, k0, lane);
    Frag b3; load_b_frag(b3, Wb, H_, col0 + 48, k0, lane);
    acc0 = __builtin_amdgcn_wmma_f32_16x16x32_bf16(false, a.v, false, b0.v, (short)0, acc0, false, false);
    acc1 = __builtin_amdgcn_wmma_f32_16x16x32_bf16(false, a.v, false, b1.v, (short)0, acc1, false, false);
    acc2 = __builtin_amdgcn_wmma_f32_16x16x32_bf16(false, a.v, false, b2.v, (short)0, acc2, false, false);
    acc3 = __builtin_amdgcn_wmma_f32_16x16x32_bf16(false, a.v, false, b3.v, (short)0, acc3, false, false);
  }

  const int nlo = lane & 15;
  const int rhi = (lane >> 4) * 8;
  #pragma unroll
  for (int r = 0; r < 8; ++r) {
    const int b = row0 + r + rhi;
    const size_t pbase = ((size_t)ks * B_ + b) * NG_;
    part[pbase + col0 +  0 + nlo] = acc0[r];
    part[pbase + col0 + 16 + nlo] = acc1[r];
    part[pbase + col0 + 32 + nlo] = acc2[r];
    part[pbase + col0 + 48 + nlo] = acc3[r];
  }
}

// ---------------- phase 2b: split-K reduction + G[t] add + LSTM cell ----------------

__global__ __launch_bounds__(256) void cell_kernel(
    const float* __restrict__ part,          // [KS_, B, 4H]
    const float* __restrict__ Gt,            // [B, 4H] pre-gates (x-proj + biases)
    float* __restrict__ c,                   // [B, H] persistent
    unsigned short* __restrict__ hb,         // [B, H] bf16 (next step A)
    float* __restrict__ out,                 // [B, T, H]
    int t)
{
  int idx = blockIdx.x * 256 + threadIdx.x;  // B*H threads
  int b = idx >> 10;
  int j = idx & (H_ - 1);
  const size_t gb = (size_t)b * NG_;

  float gi = Gt[gb + j];
  float gf = Gt[gb + H_ + j];
  float gg = Gt[gb + 2 * H_ + j];
  float go = Gt[gb + 3 * H_ + j];
  #pragma unroll
  for (int ks = 0; ks < KS_; ++ks) {
    const float* p = part + ((size_t)ks * B_) * NG_ + gb;
    gi += p[j];
    gf += p[H_ + j];
    gg += p[2 * H_ + j];
    go += p[3 * H_ + j];
  }

  float ig = 1.0f / (1.0f + __expf(-gi));
  float fg = 1.0f / (1.0f + __expf(-gf));
  float gv = tanhf(gg);
  float og = 1.0f / (1.0f + __expf(-go));
  float cn = fg * c[idx] + ig * gv;
  c[idx] = cn;
  float h = og * tanhf(cn);
  out[((size_t)b * T_ + t) * H_ + j] = h;
  hb[idx] = f2bf(h);
}

// ---------------- final: (h_f, c_f) tail of d_out ----------------

__global__ __launch_bounds__(256) void final_kernel(
    const float* __restrict__ out, const float* __restrict__ c,
    float* __restrict__ hf, float* __restrict__ cf)
{
  int idx = blockIdx.x * 256 + threadIdx.x;  // B*H threads
  int b = idx >> 10;
  int j = idx & (H_ - 1);
  hf[idx] = out[((size_t)b * T_ + (T_ - 1)) * H_ + j];
  cf[idx] = c[idx];
}

extern "C" void kernel_launch(void* const* d_in, const int* in_sizes, int n_in,
                              void* d_out, int out_size, void* d_ws, size_t ws_size,
                              hipStream_t stream) {
  (void)in_sizes; (void)n_in; (void)out_size; (void)ws_size;
  const float* x   = (const float*)d_in[0];   // [B, T, I]
  const float* h0  = (const float*)d_in[1];   // [B, H]
  const float* c0  = (const float*)d_in[2];   // [B, H]
  const float* Wih = (const float*)d_in[3];   // [4H, I]
  const float* Whh = (const float*)d_in[4];   // [4H, H]
  const float* bih = (const float*)d_in[5];   // [4H]
  const float* bhh = (const float*)d_in[6];   // [4H]
  float* out = (float*)d_out;                 // [B,T,H] ++ h_f[B,H] ++ c_f[B,H]
  float* hf  = out + (size_t)B_ * T_ * H_;
  float* cf  = hf + (size_t)B_ * H_;

  // workspace carve-out (256B aligned)
  char* ws = (char*)d_ws;
  size_t off = 0;
  auto carve = [&](size_t bytes) -> void* {
    void* p = ws + off;
    off += (bytes + 255) & ~(size_t)255;
    return p;
  };
  float*          G     = (float*)         carve((size_t)T_ * B_ * NG_ * 4);   // 512 MB
  unsigned short* x_bf  = (unsigned short*)carve((size_t)B_ * T_ * I_ * 2);    //  64 MB
  unsigned short* wih_b = (unsigned short*)carve((size_t)NG_ * I_ * 2);        //   8 MB
  unsigned short* whh_b = (unsigned short*)carve((size_t)NG_ * H_ * 2);        //   8 MB
  float*          bias  = (float*)         carve((size_t)NG_ * 4);
  float*          part  = (float*)         carve((size_t)KS_ * B_ * NG_ * 4);  //   4 MB
  float*          cbuf  = (float*)         carve((size_t)B_ * H_ * 4);
  unsigned short* hb    = (unsigned short*)carve((size_t)B_ * H_ * 2);

  // --- conversions ---
  {
    int n4 = (B_ * T_ * I_) / 4;
    cvt_bf16_kernel<<<n4 / 256, 256, 0, stream>>>((const float4*)x, (uint2*)x_bf, n4);
  }
  {
    int n4 = (NG_ * I_) / 4;
    cvt_bf16_kernel<<<n4 / 256, 256, 0, stream>>>((const float4*)Wih, (uint2*)wih_b, n4);
  }
  {
    int n4 = (NG_ * H_) / 4;
    cvt_bf16_kernel<<<n4 / 256, 256, 0, stream>>>((const float4*)Whh, (uint2*)whh_b, n4);
  }
  bias_kernel<<<NG_ / 256, 256, 0, stream>>>(bih, bhh, bias);
  init_state_kernel<<<(B_ * H_) / 256, 256, 0, stream>>>(h0, c0, hb, cbuf);

  // --- phase 1: bulk input-projection GEMM (full-device parallel) ---
  {
    dim3 grid((B_ * T_ / 16) / 8, NG_ / 64);  // (256, 64)
    gemm_x_kernel<<<grid, 256, 0, stream>>>(x_bf, wih_b, bias, G);
  }

  // --- phase 2: sequential recurrence (split-K recurrent GEMM + fused cell) ---
  for (int t = 0; t < T_; ++t) {
    const float* Gt = G + (size_t)t * B_ * NG_;
    gemm_h_kernel<<<128, 256, 0, stream>>>(hb, whh_b, part);
    cell_kernel<<<(B_ * H_) / 256, 256, 0, stream>>>(part, Gt, cbuf, hb, out, t);
  }

  final_kernel<<<(B_ * H_) / 256, 256, 0, stream>>>(out, cbuf, hf, cf);
}
